// DLinear_3367254360495
// MI455X (gfx1250) — matrix operand order repaired and checked by reading, not compiled, not run on hardware
//
#include <hip/hip_runtime.h>
#include <hip/hip_bf16.h>

// ---------------------------------------------------------------------------
// DLinear for MI455X (gfx1250): decomposition + bf16x3-split WMMA batched GEMM
//   out[b,c,p] = res[b,c,:]@W_res[c,p,:] + season[b,c,:]@W_trend[c,p,:] + bias
// Kernel 1 pre-splits res/season into bf16 hi/lo planes (WMMA-A-ready).
// Kernel 2: per-channel GEMM; K chunks staged by the Tensor Data Mover,
// double-buffered (s_wait_tensorcnt + barriers); weights split to bf16 hi/lo
// once per workgroup, in place in LDS; fragments unpacked with v_perm_b32.
// ---------------------------------------------------------------------------

typedef __attribute__((ext_vector_type(16))) __bf16        v16bf;
typedef __attribute__((ext_vector_type(8)))  float         v8f;
typedef __attribute__((ext_vector_type(4)))  unsigned int  v4u;
typedef __attribute__((ext_vector_type(8)))  int           v8i;
typedef __attribute__((ext_vector_type(4)))  int           v4i;

#define C_CH   321
#define BATCH  64
#define SEQ    720
#define PRED   336
#define WIN    25
#define PADW   12
#define SPAD   736      // SEQ padded to 23*32
#define KC     32       // K per chunk (bf16 WMMA K)
#define NKC    23       // number of K chunks
#define BN     48       // N per workgroup (3 x 16)

#if __has_builtin(__builtin_amdgcn_tensor_load_to_lds)
#define HAVE_TDM 1
#else
#define HAVE_TDM 0
#endif

// Split fp32 into bf16 hi (truncated top 16 bits) + bf16 lo (RNE of remainder).
__device__ __forceinline__ void splitf(float x, unsigned& h, unsigned& l) {
  unsigned u  = __float_as_uint(x);
  unsigned hm = u & 0xFFFF0000u;
  h = u >> 16;
  float r = x - __uint_as_float(hm);
  unsigned ur = __float_as_uint(r);
  ur += 0x7FFFu + ((ur >> 16) & 1u);   // round-to-nearest-even to bf16
  l = ur >> 16;
}

// Pack fp32 as (hi_bf16 << 16) | lo_bf16 in one dword.
__device__ __forceinline__ unsigned packbf(float v) {
  unsigned h, l;
  splitf(v, h, l);
  return (h << 16) | l;
}

__device__ __forceinline__ v8f wmma_bf16(v16bf a, v16bf b, v8f c) {
  return __builtin_amdgcn_wmma_f32_16x16x32_bf16(
      /*neg_a=*/false, a, /*neg_b=*/false, b,
      /*c_mod=*/(short)0, c, /*reuse_a=*/false, /*reuse_b=*/false);
}

#if HAVE_TDM
// 2D TDM load: tile = KC (dim0, contiguous) x tile_rows rows, element size
// 2^dsz bytes (dsz=1: 2B, dsz=2: 4B). tensor_dim0 < KC zero-pads the K tail.
// 6-arg builtin (clang-23): (v4u g0, v8i g1, v4i g2, v4i g3, v8i g4, i32 cpol)
__device__ __forceinline__ void tdm_load_2d(unsigned lds_off, const void* gptr,
                                            unsigned dsz,
                                            unsigned tdim0, unsigned rows,
                                            unsigned tile_rows,
                                            unsigned long long stride) {
  unsigned long long ga = (unsigned long long)(uintptr_t)gptr;
  v4u g0;
  g0[0] = 1u;                                   // count=1, user descriptor
  g0[1] = lds_off;                              // LDS byte address
  g0[2] = (unsigned)ga;                         // global_addr[31:0]
  g0[3] = ((unsigned)(ga >> 32) & 0x01FFFFFFu)  // global_addr[56:32]
        | (2u << 30);                           // type = 2 ("image")
  v8i g1;
  g1[0] = (int)(dsz << 16);                             // data_size
  g1[1] = (int)((tdim0 & 0xFFFFu) << 16);               // tensor_dim0 lo
  g1[2] = (int)((tdim0 >> 16) | ((rows & 0xFFFFu) << 16)); // tensor_dim1 lo
  g1[3] = (int)((rows >> 16) | ((unsigned)KC << 16));   // tile_dim0 = 32
  g1[4] = (int)(tile_rows & 0xFFFFu);                   // tile_dim1
  g1[5] = (int)(stride & 0xFFFFFFFFull);                // tensor_dim0_stride
  g1[6] = (int)((stride >> 32) & 0xFFFFull);
  g1[7] = 0;
  v4i z4 = {0, 0, 0, 0};                                // 2D: groups 2/3 unused
  v8i z8 = {0, 0, 0, 0, 0, 0, 0, 0};
  __builtin_amdgcn_tensor_load_to_lds(g0, g1, z4, z4, z8, 0);
}
#endif

// ---------------------------------------------------------------------------
// Kernel 1: series decomposition + bf16 hi/lo pre-split.
// x[B,S,C] -> 4 ushort planes [C][B][SPAD]: resHi, resLo, seaHi, seaLo.
// AvgPool1d(k=25, pad=12, count_include_pad=True): zero-padded sum / 25.
// Stores are dword-packed pairs of adjacent s.
// ---------------------------------------------------------------------------
__global__ void __launch_bounds__(256)
dlinear_decomp(const float* __restrict__ x,
               unsigned* __restrict__ resHiD, unsigned* __restrict__ resLoD,
               unsigned* __restrict__ seaHiD, unsigned* __restrict__ seaLoD) {
  __shared__ float tile[32][160];            // 152 used; padded stride
  const int s0  = blockIdx.x * 128;          // 6 chunks of 128 along S
  const int c0  = blockIdx.y * 32;           // 11 channel tiles
  const int b   = blockIdx.z;
  const int tid = threadIdx.x;

  // Phase 1: stage x[b, s0-12 .. s0+139, c0..c0+31] (zero outside [0,SEQ)).
  {
    const int c    = tid & 31;
    const int srow = tid >> 5;               // 0..7
    const int cc   = c0 + c;
#pragma unroll
    for (int j = 0; j < 19; ++j) {
      int idx = srow + 8 * j;
      if (idx < 152) {
        int sg = s0 - PADW + idx;
        float v = 0.0f;
        if (cc < C_CH && sg >= 0 && sg < SEQ)
          v = x[((size_t)b * SEQ + sg) * C_CH + cc];
        tile[c][idx] = v;
      }
    }
  }
  __syncthreads();

  // Phase 2: 25-tap mean (incremental for pair), residual, split, store.
  {
    const int c    = tid >> 3;               // 0..31
    const int pidx = tid & 7;                // pair index
    const int cc   = c0 + c;
    if (cc < C_CH) {
      const size_t rowOff = ((size_t)cc * BATCH + b) * SPAD;  // even
#pragma unroll
      for (int j = 0; j < 8; ++j) {
        int srel = pidx * 2 + 16 * j;        // 0,2,...,126
        int s    = s0 + srel;                // even; SEQ=720 even
        if (s >= SPAD) continue;
        unsigned rh = 0u, rl = 0u, sh = 0u, sl = 0u;
        if (s < SEQ) {
          float sum0 = 0.0f;
#pragma unroll
          for (int t = 0; t < WIN; ++t) sum0 += tile[c][srel + t];
          float sum1 = sum0 - tile[c][srel] + tile[c][srel + WIN];
          float sea0 = sum0 * (1.0f / (float)WIN);
          float sea1 = sum1 * (1.0f / (float)WIN);
          float rs0  = tile[c][srel + PADW] - sea0;
          float rs1  = tile[c][srel + PADW + 1] - sea1;
          unsigned h0, l0, h1, l1;
          splitf(rs0, h0, l0); splitf(rs1, h1, l1);
          rh = h0 | (h1 << 16); rl = l0 | (l1 << 16);
          splitf(sea0, h0, l0); splitf(sea1, h1, l1);
          sh = h0 | (h1 << 16); sl = l0 | (l1 << 16);
        }
        const size_t d = (rowOff + (size_t)s) >> 1;
        resHiD[d] = rh; resLoD[d] = rl;
        seaHiD[d] = sh; seaLoD[d] = sl;
      }
    }
  }
}

// ---------------------------------------------------------------------------
// Kernel 2: per-channel GEMM, M=64, N=48/block, K=736 (chunks of 32).
// 4 waves; wave w owns M-tile w, iterates 3 N-tiles. bf16x3 split per chain,
// both chains (res@Wres, season@Wtrend) into one f32 accumulator.
// ---------------------------------------------------------------------------
__global__ void __launch_bounds__(128)
dlinear_gemm(const unsigned short* __restrict__ aRH,
             const unsigned short* __restrict__ aRL,
             const unsigned short* __restrict__ aSH,
             const unsigned short* __restrict__ aSL,
             const float* __restrict__ Wres, const float* __restrict__ Wtrd,
             const float* __restrict__ br,   const float* __restrict__ bt,
             float* __restrict__ out) {
  // A planes: bf16, WMMA-ready rows of KC elements.
  __shared__ __align__(16) unsigned short sARH[2][BATCH * KC];
  __shared__ __align__(16) unsigned short sARL[2][BATCH * KC];
  __shared__ __align__(16) unsigned short sASH[2][BATCH * KC];
  __shared__ __align__(16) unsigned short sASL[2][BATCH * KC];
  // B tiles: TDM lands raw fp32; converted in place to (hi<<16)|lo dwords.
  __shared__ __align__(16) unsigned sBres[2][BN * KC];
  __shared__ __align__(16) unsigned sBtrd[2][BN * KC];

  const int n0   = blockIdx.x * BN;          // 0,48,...,288
  const int ch   = blockIdx.y;               // channel
  const int tid  = threadIdx.x;
  const int wave = tid >> 5;
  const int lane = tid & 31;
  const int l15  = lane & 15;
  const int hl   = lane >> 4;                // half-wave select

  const size_t aOff = (size_t)ch * BATCH * SPAD;
  const unsigned short* aRHg = aRH + aOff;
  const unsigned short* aRLg = aRL + aOff;
  const unsigned short* aSHg = aSH + aOff;
  const unsigned short* aSLg = aSL + aOff;
  const float* bResG = Wres + ((size_t)ch * PRED + n0) * SEQ;
  const float* bTrdG = Wtrd + ((size_t)ch * PRED + n0) * SEQ;

  v8f acc[3];
#pragma unroll
  for (int n = 0; n < 3; ++n) acc[n] = (v8f){0.f,0.f,0.f,0.f,0.f,0.f,0.f,0.f};

  auto issue = [&](int kk, int buf) {
    const unsigned k0 = (unsigned)kk * KC;
#if HAVE_TDM
    if (wave == 0) {
      tdm_load_2d((unsigned)(uintptr_t)&sARH[buf][0], aRHg + k0, 1u,
                  SPAD - k0, BATCH, BATCH, SPAD);
      tdm_load_2d((unsigned)(uintptr_t)&sARL[buf][0], aRLg + k0, 1u,
                  SPAD - k0, BATCH, BATCH, SPAD);
      tdm_load_2d((unsigned)(uintptr_t)&sASH[buf][0], aSHg + k0, 1u,
                  SPAD - k0, BATCH, BATCH, SPAD);
      tdm_load_2d((unsigned)(uintptr_t)&sASL[buf][0], aSLg + k0, 1u,
                  SPAD - k0, BATCH, BATCH, SPAD);
      // tensor_dim0 = SEQ-k0: TDM zero-fills K tail of the last chunk.
      tdm_load_2d((unsigned)(uintptr_t)&sBres[buf][0], bResG + k0, 2u,
                  SEQ - k0, BN, BN, SEQ);
      tdm_load_2d((unsigned)(uintptr_t)&sBtrd[buf][0], bTrdG + k0, 2u,
                  SEQ - k0, BN, BN, SEQ);
    }
#else
    // Cooperative fallback copies (dword granularity for the A planes).
    const unsigned* g0 = (const unsigned*)aRHg;
    const unsigned* g1 = (const unsigned*)aRLg;
    const unsigned* g2 = (const unsigned*)aSHg;
    const unsigned* g3 = (const unsigned*)aSLg;
    for (int i = tid; i < BATCH * KC / 2; i += 128) {
      int r = i >> 4, kd = i & 15;
      size_t gi = (size_t)r * (SPAD / 2) + (k0 / 2) + kd;
      ((unsigned*)&sARH[buf][0])[i] = g0[gi];
      ((unsigned*)&sARL[buf][0])[i] = g1[gi];
      ((unsigned*)&sASH[buf][0])[i] = g2[gi];
      ((unsigned*)&sASL[buf][0])[i] = g3[gi];
    }
    for (int i = tid; i < BN * KC; i += 128) {
      int r = i >> 5, kq = i & 31;
      unsigned ks = k0 + kq;
      float fr = (ks < SEQ) ? bResG[(size_t)r * SEQ + ks] : 0.0f;
      float ft = (ks < SEQ) ? bTrdG[(size_t)r * SEQ + ks] : 0.0f;
      sBres[buf][i] = __float_as_uint(fr);
      sBtrd[buf][i] = __float_as_uint(ft);
    }
#endif
  };

  issue(0, 0);

  for (int k = 0; k < NKC; ++k) {
    const int cur = k & 1;
    if (k + 1 < NKC) issue(k + 1, cur ^ 1);
#if HAVE_TDM
    if (wave == 0) {
      if (k + 1 < NKC) __builtin_amdgcn_s_wait_tensorcnt((short)6);
      else             __builtin_amdgcn_s_wait_tensorcnt((short)0);
    }
#endif
    __syncthreads();   // chunk k raw data resident in LDS

    // ---- Cooperative in-place weight split: fp32 -> (hi<<16)|lo dwords ----
#pragma unroll
    for (int i = tid; i < BN * KC; i += 128) {
      sBres[cur][i] = packbf(__uint_as_float(sBres[cur][i]));
      sBtrd[cur][i] = packbf(__uint_as_float(sBtrd[cur][i]));
    }
    __syncthreads();   // packed weights visible to all waves

    // ---- A fragments (ISA A layout: lanes<16 K{0..7,16..23}, else +8) ----
    const int arow = (wave * 16 + l15) * KC + (hl ? 8 : 0);
    union BV { v16bf v; uint4 q[2]; unsigned u[8]; };
    BV fARH, fARL, fASH, fASL;
    fARH.q[0] = *(const uint4*)(&sARH[cur][arow]);
    fARH.q[1] = *(const uint4*)(&sARH[cur][arow + 16]);
    fARL.q[0] = *(const uint4*)(&sARL[cur][arow]);
    fARL.q[1] = *(const uint4*)(&sARL[cur][arow + 16]);
    fASH.q[0] = *(const uint4*)(&sASH[cur][arow]);
    fASH.q[1] = *(const uint4*)(&sASH[cur][arow + 16]);
    fASL.q[0] = *(const uint4*)(&sASL[cur][arow]);
    fASL.q[1] = *(const uint4*)(&sASL[cur][arow + 16]);

    // ---- 3 N-tiles: unpack packed weights with v_perm, 6 WMMAs each ----
    const int kb2 = hl ? 8 : 0;
#pragma unroll
    for (int n = 0; n < 3; ++n) {
      const unsigned* bR = &sBres[cur][(n * 16 + l15) * KC + kb2];
      const unsigned* bT = &sBtrd[cur][(n * 16 + l15) * KC + kb2];
      uint4 tb4[4], tt4[4];
      tb4[0] = *(const uint4*)(bR + 0);  tb4[1] = *(const uint4*)(bR + 4);
      tb4[2] = *(const uint4*)(bR + 16);  tb4[3] = *(const uint4*)(bR + 20);
      tt4[0] = *(const uint4*)(bT + 0);  tt4[1] = *(const uint4*)(bT + 4);
      tt4[2] = *(const uint4*)(bT + 16);  tt4[3] = *(const uint4*)(bT + 20);
      const unsigned* tb = (const unsigned*)tb4;
      const unsigned* tt = (const unsigned*)tt4;

      BV bRH, bRL, bTH, bTL;
#pragma unroll
      for (int j = 0; j < 8; ++j) {
        unsigned p0 = tb[2 * j], p1 = tb[2 * j + 1];
        bRH.u[j] = __builtin_amdgcn_perm(p1, p0, 0x07060302u);  // hi halves
        bRL.u[j] = __builtin_amdgcn_perm(p1, p0, 0x05040100u);  // lo halves
        unsigned q0 = tt[2 * j], q1 = tt[2 * j + 1];
        bTH.u[j] = __builtin_amdgcn_perm(q1, q0, 0x07060302u);
        bTL.u[j] = __builtin_amdgcn_perm(q1, q0, 0x05040100u);
      }

      v8f a = acc[n];
      a = wmma_bf16(fARH.v, bRH.v, a);   // res  @ Wres  : hi*hi
      a = wmma_bf16(fARL.v, bRH.v, a);   //               lo*hi
      a = wmma_bf16(fARH.v, bRL.v, a);   //               hi*lo
      a = wmma_bf16(fASH.v, bTH.v, a);   // seas @ Wtrend: hi*hi
      a = wmma_bf16(fASL.v, bTH.v, a);   //               lo*hi
      a = wmma_bf16(fASH.v, bTL.v, a);   //               hi*lo
      acc[n] = a;
    }
    __syncthreads();   // all waves done with `cur` before TDM reuses it
  }

  // ---- Epilogue: bias + store out[B,C,P] (C/D layout: VGPR r -> M=r / r+8) -
#pragma unroll
  for (int n = 0; n < 3; ++n) {
    const int p = n0 + n * 16 + l15;
    const float bias = br[(size_t)ch * PRED + p] + bt[(size_t)ch * PRED + p];
#pragma unroll
    for (int r = 0; r < 8; ++r) {
      const int m = wave * 16 + hl * 8 + r;
      out[((size_t)m * C_CH + ch) * PRED + p] = acc[n][r] + bias;
    }
  }
}

// ---------------------------------------------------------------------------
extern "C" void kernel_launch(void* const* d_in, const int* in_sizes, int n_in,
                              void* d_out, int out_size, void* d_ws,
                              size_t ws_size, hipStream_t stream) {
  (void)in_sizes; (void)n_in; (void)out_size; (void)ws_size;
  const float* x    = (const float*)d_in[0];
  const float* Wres = (const float*)d_in[1];
  const float* brs  = (const float*)d_in[2];
  const float* Wtrd = (const float*)d_in[3];
  const float* btr  = (const float*)d_in[4];
  float* out = (float*)d_out;

  const size_t plane = (size_t)C_CH * BATCH * SPAD;      // ushort elements
  unsigned short* pRH = (unsigned short*)d_ws;
  unsigned short* pRL = pRH + plane;
  unsigned short* pSH = pRL + plane;
  unsigned short* pSL = pSH + plane;

  dim3 g1(6, 11, BATCH);                 // S chunks x channel tiles x batch
  dlinear_decomp<<<g1, 256, 0, stream>>>(x, (unsigned*)pRH, (unsigned*)pRL,
                                         (unsigned*)pSH, (unsigned*)pSL);

  dim3 g2(PRED / BN, C_CH);              // 7 N-blocks x 321 channels
  dlinear_gemm<<<g2, 128, 0, stream>>>(pRH, pRL, pSH, pSL,
                                       Wres, Wtrd, brs, btr, out);
}
